// LabelWiseAttentionV1_84774064488622
// MI455X (gfx1250) — matrix operand order repaired and checked by reading, not compiled, not run on hardware
//
#include <hip/hip_runtime.h>

#define B_ 8
#define S_ 2048
#define D_ 1024
#define L_ 4096

typedef unsigned short ushort_t;
typedef __attribute__((ext_vector_type(16))) __bf16 v16bf;
typedef __attribute__((ext_vector_type(8)))  __bf16 v8bf;
typedef __attribute__((ext_vector_type(8)))  float  v8f;

// float -> bf16 round-to-nearest-even (bit trick; inputs are well-behaved normals)
__device__ __forceinline__ ushort_t f2bf(float f) {
    unsigned u = __float_as_uint(f);
    unsigned r = u + 0x7FFFu + ((u >> 16) & 1u);
    return (ushort_t)(r >> 16);
}

union AFrag { v16bf v; v8bf h[2]; };

// A-matrix fragment, row-major [M x K] bf16, 16x32 tile at (m_base, k0).
// ISA layout: lane<16 holds row (lane), halfs = K{k0..k0+7, k0+16..k0+23};
//             lane>=16 holds row (lane-16), halfs = K{k0+8.., k0+24..}.
__device__ __forceinline__ v16bf load_a_frag(const ushort_t* A, int lda,
                                             int m_base, int k0, int lane) {
    const int m  = m_base + (lane & 15);
    const int kk = k0 + ((lane >> 4) << 3);          // +8 for upper half-wave
    const v8bf* p = (const v8bf*)(A + (size_t)m * lda + kk);
    AFrag a; a.h[0] = p[0]; a.h[1] = p[2];           // k..k+7 and k+16..k+23
    return a.v;
}

// B-matrix fragment for NT gemm: Bt is row-major [N x K] bf16 (i.e. B^T).
// 32x16 tile: lane n (0..15) = column n, holds K{k0..k0+7, k0+16..k0+23} (lane>=16: +8).
__device__ __forceinline__ v16bf load_bt_frag(const ushort_t* Bt, int ldb,
                                              int n_base, int k0, int lane) {
    const int n  = n_base + (lane & 15);
    const int kk = k0 + ((lane >> 4) << 3);          // +8 for upper half-wave
    const v8bf* p = (const v8bf*)(Bt + (size_t)n * ldb + kk);
    AFrag b; b.h[0] = p[0]; b.h[1] = p[2];
    return b.v;
}

// ---------------- prep kernels ----------------

__global__ void cvt_mask_kernel(const float* __restrict__ h, const float* __restrict__ msk,
                                ushort_t* __restrict__ o, int total) {
    int i = blockIdx.x * blockDim.x + threadIdx.x;
    if (i < total) o[i] = f2bf(h[i] * msk[i / D_]);   // msk indexed by (b*S+s)
}

__global__ void cvt_kernel(const float* __restrict__ in, ushort_t* __restrict__ o, int total) {
    int i = blockIdx.x * blockDim.x + threadIdx.x;
    if (i < total) o[i] = f2bf(in[i]);
}

__global__ void transpose_cvt_kernel(const float* __restrict__ in, ushort_t* __restrict__ o) {
    int i = blockIdx.x * blockDim.x + threadIdx.x;   // i = d*D + e
    if (i < D_ * D_) {
        int d = i / D_, e = i % D_;
        o[i] = f2bf(in[(size_t)e * D_ + d]);
    }
}

__global__ void cv_kernel(const float* __restrict__ O, const float* __restrict__ bv,
                          float* __restrict__ cv) {
    int l = blockIdx.x * blockDim.x + threadIdx.x;
    if (l < L_) {
        float s = 0.f;
        for (int e = 0; e < D_; ++e) s += O[(size_t)l * D_ + e] * bv[e];
        cv[l] = s;
    }
}

// ---------------- NT gemm: C[MxN] = A[MxK] * Bt[NxK]^T (bf16 in/out, f32 acc) ----------------

__global__ __launch_bounds__(128) void gemm_nt_bf16(const ushort_t* __restrict__ A,
                                                    const ushort_t* __restrict__ Bt,
                                                    ushort_t* __restrict__ C,
                                                    int M, int N, int K) {
    const int lane   = threadIdx.x & 31;
    const int w      = threadIdx.x >> 5;
    const int m_base = blockIdx.x * 64 + w * 16;
    const int n_base = blockIdx.y * 64;

    v8f acc0 = {}, acc1 = {}, acc2 = {}, acc3 = {};
    for (int k0 = 0; k0 < K; k0 += 32) {
        // distinct registers for all fragments -> loads clause together,
        // WMMAs issue back-to-back while next loads are in flight
        v16bf a  = load_a_frag(A, K, m_base, k0, lane);
        v16bf b0 = load_bt_frag(Bt, K, n_base,      k0, lane);
        v16bf b1 = load_bt_frag(Bt, K, n_base + 16, k0, lane);
        v16bf b2 = load_bt_frag(Bt, K, n_base + 32, k0, lane);
        v16bf b3 = load_bt_frag(Bt, K, n_base + 48, k0, lane);
        acc0 = __builtin_amdgcn_wmma_f32_16x16x32_bf16(false, a, false, b0, (short)0, acc0, false, false);
        acc1 = __builtin_amdgcn_wmma_f32_16x16x32_bf16(false, a, false, b1, (short)0, acc1, false, false);
        acc2 = __builtin_amdgcn_wmma_f32_16x16x32_bf16(false, a, false, b2, (short)0, acc2, false, false);
        acc3 = __builtin_amdgcn_wmma_f32_16x16x32_bf16(false, a, false, b3, (short)0, acc3, false, false);
    }
    v8f acc[4] = {acc0, acc1, acc2, acc3};
    #pragma unroll
    for (int t = 0; t < 4; ++t)
        #pragma unroll
        for (int r = 0; r < 8; ++r) {
            int m = m_base + r + (lane >> 4) * 8;
            int n = n_base + t * 16 + (lane & 15);
            C[(size_t)m * N + n] = f2bf(acc[t][r]);
        }
}

// ---------------- async staging helpers (CDNA5 ASYNCcnt path) ----------------

// Issue wave-level async copies: 64 KB chunk (32 rows x D_ bf16) global -> LDS.
// Each executed instruction moves 32 lanes x 16B = 512B with no VGPR landing.
__device__ __forceinline__ void async_stage(const ushort_t* __restrict__ gsrc,
                                            ushort_t* lbase, int tid) {
    // low 32 bits of a generic shared-aperture address are the LDS byte offset
    unsigned lds0 = (unsigned)(unsigned long long)(lbase);
    #pragma unroll 8
    for (int i = tid; i < (32 * D_) / 8; i += 128) {
        unsigned loff = lds0 + (unsigned)i * 16u;
        const v8bf* p = (const v8bf*)gsrc + i;
        asm volatile("global_load_async_to_lds_b128 %0, %1, off"
                     :: "v"(loff), "v"(p) : "memory");
    }
}

__device__ __forceinline__ void wait_async0() {
    asm volatile("s_wait_asynccnt 0x0" ::: "memory");
}

// ---------------- fused dual-gemm + online softmax ----------------
// block = 4 waves; tile = 64 labels x all S (streamed in 32-wide chunks), K = D.
// h chunks double-buffered in LDS via global_load_async_to_lds_b128.

__global__ __launch_bounds__(128) void lwan_kernel(const ushort_t* __restrict__ hbf, // [B,S,D]
                                                   const ushort_t* __restrict__ Ek,  // [L,D]
                                                   const ushort_t* __restrict__ Ov,  // [L,D]
                                                   const float* __restrict__ cv,     // [L]
                                                   float* __restrict__ out) {        // [B,L]
    __shared__ ushort_t hs[2][32 * D_];              // 2 x 64 KB double buffer
    const int b      = blockIdx.y;
    const int Lbase  = blockIdx.x * 64;
    const int tid    = threadIdx.x;
    const int lane   = tid & 31;
    const int w      = tid >> 5;
    const int m_base = Lbase + w * 16;

    float mrow[8], Zp[8], Np[8];
    #pragma unroll
    for (int r = 0; r < 8; ++r) { mrow[r] = -3.0e38f; Zp[r] = 0.f; Np[r] = 0.f; }

    const ushort_t* hB = hbf + (size_t)b * S_ * D_;

    // prologue: stage chunk 0 into buffer 0
    async_stage(hB, hs[0], tid);

    const int NCH = S_ / 32;
    for (int ci = 0; ci < NCH; ++ci) {
        const int cur = ci & 1;
        wait_async0();                               // this wave's staged data is in LDS
        __syncthreads();                             // all waves staged + done reading prev buf
        if (ci + 1 < NCH)                            // overlap next chunk's copy with compute
            async_stage(hB + (size_t)(ci + 1) * 32 * D_, hs[1 - cur], tid);

        v8f sc0 = {}, sc1 = {}, vd0 = {}, vd1 = {};
        #pragma unroll 4
        for (int k0 = 0; k0 < D_; k0 += 32) {
            v16bf aE = load_a_frag(Ek, D_, m_base, k0, lane);
            v16bf aO = load_a_frag(Ov, D_, m_base, k0, lane);
            v16bf b0 = load_bt_frag(hs[cur], D_, 0,  k0, lane);  // s = s0+0..15
            v16bf b1 = load_bt_frag(hs[cur], D_, 16, k0, lane);  // s = s0+16..31
            sc0 = __builtin_amdgcn_wmma_f32_16x16x32_bf16(false, aE, false, b0, (short)0, sc0, false, false);
            sc1 = __builtin_amdgcn_wmma_f32_16x16x32_bf16(false, aE, false, b1, (short)0, sc1, false, false);
            vd0 = __builtin_amdgcn_wmma_f32_16x16x32_bf16(false, aO, false, b0, (short)0, vd0, false, false);
            vd1 = __builtin_amdgcn_wmma_f32_16x16x32_bf16(false, aO, false, b1, (short)0, vd1, false, false);
        }

        // Online softmax update. C-tile layout: VGPR r, lanes 0-15 -> row r,
        // lanes 16-31 -> row r+8; N = lane&15. Row reductions = width-16 shuffles.
        #pragma unroll
        for (int r = 0; r < 8; ++r) {
            float t = fmaxf(sc0[r], sc1[r]);
            #pragma unroll
            for (int off = 1; off < 16; off <<= 1)
                t = fmaxf(t, __shfl_xor(t, off, 16));
            float mo = mrow[r];
            float mn = fmaxf(mo, t);
            float scale = __expf(mo - mn);
            float p0 = __expf(sc0[r] - mn);
            float p1 = __expf(sc1[r] - mn);
            Zp[r] = Zp[r] * scale + p0 + p1;
            Np[r] = Np[r] * scale + p0 * vd0[r] + p1 * vd1[r];
            mrow[r] = mn;
        }
    }

    #pragma unroll
    for (int r = 0; r < 8; ++r) {
        float z = Zp[r], n = Np[r];
        #pragma unroll
        for (int off = 1; off < 16; off <<= 1) {
            z += __shfl_xor(z, off, 16);
            n += __shfl_xor(n, off, 16);
        }
        if ((lane & 15) == 0) {
            int l = m_base + r + (lane >> 4) * 8;
            out[(size_t)b * L_ + l] = n / z + cv[l];
        }
    }
}

// ---------------- launch ----------------

extern "C" void kernel_launch(void* const* d_in, const int* in_sizes, int n_in,
                              void* d_out, int out_size, void* d_ws, size_t ws_size,
                              hipStream_t stream) {
    const float* h   = (const float*)d_in[0];
    const float* msk = (const float*)d_in[1];
    const float* Wk  = (const float*)d_in[2];
    const float* bk  = (const float*)d_in[3]; (void)bk; // cancels in softmax
    const float* Wv  = (const float*)d_in[4];
    const float* bv  = (const float*)d_in[5];
    const float* E   = (const float*)d_in[6];
    const float* O   = (const float*)d_in[7];
    float* out = (float*)d_out;
    (void)in_sizes; (void)n_in; (void)out_size; (void)ws_size;

    char* ws = (char*)d_ws;
    size_t off = 0;
    auto alloc = [&](size_t bytes) -> void* {
        void* p = ws + off;
        off += (bytes + 255) & ~(size_t)255;
        return p;
    };
    ushort_t* hbf = (ushort_t*)alloc((size_t)B_ * S_ * D_ * 2);  // 32 MB
    ushort_t* Ebf = (ushort_t*)alloc((size_t)L_ * D_ * 2);       // 8 MB
    ushort_t* Obf = (ushort_t*)alloc((size_t)L_ * D_ * 2);       // 8 MB
    ushort_t* WkT = (ushort_t*)alloc((size_t)D_ * D_ * 2);       // 2 MB
    ushort_t* WvT = (ushort_t*)alloc((size_t)D_ * D_ * 2);       // 2 MB
    ushort_t* Ek  = (ushort_t*)alloc((size_t)L_ * D_ * 2);       // 8 MB
    ushort_t* Ov  = (ushort_t*)alloc((size_t)L_ * D_ * 2);       // 8 MB
    float*    cv  = (float*)alloc((size_t)L_ * 4);

    const int HT = B_ * S_ * D_;
    cvt_mask_kernel<<<(HT + 255) / 256, 256, 0, stream>>>(h, msk, hbf, HT);
    cvt_kernel<<<(L_ * D_ + 255) / 256, 256, 0, stream>>>(E, Ebf, L_ * D_);
    cvt_kernel<<<(L_ * D_ + 255) / 256, 256, 0, stream>>>(O, Obf, L_ * D_);
    transpose_cvt_kernel<<<(D_ * D_ + 255) / 256, 256, 0, stream>>>(Wk, WkT);
    transpose_cvt_kernel<<<(D_ * D_ + 255) / 256, 256, 0, stream>>>(Wv, WvT);
    cv_kernel<<<(L_ + 255) / 256, 256, 0, stream>>>(O, bv, cv);

    // Ek = E @ Wk, Ov = O @ Wv  (NT gemms against transposed weights)
    gemm_nt_bf16<<<dim3(L_ / 64, D_ / 64), 128, 0, stream>>>(Ebf, WkT, Ek, L_, D_, D_);
    gemm_nt_bf16<<<dim3(L_ / 64, D_ / 64), 128, 0, stream>>>(Obf, WvT, Ov, L_, D_, D_);

    // Fused dual-gemm + online softmax + weighted reduction
    lwan_kernel<<<dim3(L_ / 64, B_), 128, 0, stream>>>(hbf, Ek, Ov, cv, out);
}
